// ExtendedEncoderLayer_81758997446953
// MI455X (gfx1250) — matrix operand order repaired and checked by reading, not compiled, not run on hardware
//
#include <hip/hip_runtime.h>

// ======================= CDNA5 (gfx1250) encoder layer =======================
// wave32, WMMA bf16 (v_wmma_f32_16x16x32_bf16), f32 accumulation.
// Relative attention uses the qE factorization (q @ Ek^T gathered by rel id)
// so the [S,S,64] Erel tensor and the [B,H,S,S] score tensor never hit HBM.

typedef __attribute__((ext_vector_type(16))) __bf16         v16bf;
typedef __attribute__((ext_vector_type(8)))  float          v8f;
typedef __attribute__((ext_vector_type(16))) unsigned short u16x16;
typedef unsigned short u16;

#define B_   4
#define S_   1024
#define D_   1024
#define H_   16
#define DH_  64
#define FFN_ 4096
#define R_   64
#define TOK_ (B_ * S_)

// ----------------------------- helpers --------------------------------------

__device__ __forceinline__ u16 f2bf(float f) {           // f32 -> bf16 (RNE)
  unsigned u = __builtin_bit_cast(unsigned, f);
  unsigned r = u + 0x7FFFu + ((u >> 16) & 1u);
  return (u16)(r >> 16);
}

__device__ __forceinline__ v8f wmma_bf16(v16bf a, v16bf b, v8f c) {
  // (neg_a, A, neg_b, B, c_mod, C, reuse_a, reuse_b)
  return __builtin_amdgcn_wmma_f32_16x16x32_bf16(false, a, false, b,
                                                 (short)0, c, false, false);
}

// A fragment: 16x32 tile of row-major bf16 matrix at (m0,k0), leading dim lda.
// lane m=l%16; half=l/16; elem j -> K=half*16+j (contiguous, matches B).
__device__ __forceinline__ v16bf load_a_frag(const u16* A, int lda, int m0,
                                             int k0, int lane) {
  const int m  = m0 + (lane & 15);
  const int kb = k0 + ((lane >> 4) << 4);
  const u16* p = A + (size_t)m * lda + kb;
  u16x16 v;
#pragma unroll
  for (int j = 0; j < 8; ++j) { v[j] = p[j]; v[j + 8] = p[8 + j]; }
  return __builtin_bit_cast(v16bf, v);
}

// B fragment whose source memory is the transpose (row-major [N x ldt]):
// contiguous 16-element run per lane (used for K^T, V^T and Ek^T).
__device__ __forceinline__ v16bf load_bT_frag(const u16* T, int ldt, int n0,
                                              int k0, int lane) {
  const int n  = n0 + (lane & 15);
  const int kb = k0 + ((lane >> 4) << 4);
  const u16* p = T + (size_t)n * ldt + kb;
  u16x16 v;
#pragma unroll
  for (int i = 0; i < 16; ++i) v[i] = p[i];
  return __builtin_bit_cast(v16bf, v);
}

__device__ __forceinline__ float wave_max(float v) {
#pragma unroll
  for (int o = 16; o > 0; o >>= 1) v = fmaxf(v, __shfl_xor(v, o, 32));
  return v;
}
__device__ __forceinline__ float wave_sum(float v) {
#pragma unroll
  for (int o = 16; o > 0; o >>= 1) v += __shfl_xor(v, o, 32);
  return v;
}

// --------------------------- f32 -> bf16 convert -----------------------------

__global__ __launch_bounds__(256) void cvt_kernel(const float* __restrict__ s,
                                                  u16* __restrict__ d, int n) {
  for (int i = blockIdx.x * blockDim.x + threadIdx.x; i < n;
       i += gridDim.x * blockDim.x)
    d[i] = f2bf(s[i]);
}

// ----------------- V transpose: v[b,s,h*64+d] -> vT[b,h,d,s] -----------------

__global__ __launch_bounds__(256) void vtrans_kernel(const u16* __restrict__ v,
                                                     u16* __restrict__ vT) {
  for (int i = blockIdx.x * blockDim.x + threadIdx.x; i < TOK_ * D_;
       i += gridDim.x * blockDim.x) {
    const int b = i >> 20;            // / (S_*D_)
    const int s = (i >> 10) & 1023;   // row
    const int c = i & 1023;           // col = h*64+d
    const int h = c >> 6;
    const int d = c & 63;
    vT[(((size_t)b * H_ + h) * DH_ + d) * S_ + s] = v[i];
  }
}

// ------------------------------ generic GEMM ---------------------------------
// C[M,N] = A[M,K] @ B[K,N] + bias (RELU/OUTF/OUTB compile-time). Block tile
// 128x128, 8 waves as 4(M) x 2(N), wave tile 32x64 (2x4 WMMA accumulators).
// B tile staged transposed in LDS (double-buffered -> 1 barrier per K-step),
// with global_prefetch of the tile-after-next.

template <bool RELU, bool OUTF, bool OUTB>
__global__ __launch_bounds__(256) void gemm_bf16_kernel(
    const u16* __restrict__ A, int lda, const u16* __restrict__ Bm, int ldb,
    const float* __restrict__ bias, float* __restrict__ Cf,
    u16* __restrict__ Cb, int ldc, int K) {
  __shared__ u16 Bt[2][128][33];  // [buf][n][k], padded

  const int tid  = threadIdx.x;
  const int lane = tid & 31;
  const int wave = tid >> 5;
  const int wm   = wave >> 1;          // 0..3
  const int wn   = wave & 1;           // 0..1
  const int m0   = blockIdx.y * 128 + wm * 32;
  const int n0   = blockIdx.x * 128;

  const v8f vzero = {0.f, 0.f, 0.f, 0.f, 0.f, 0.f, 0.f, 0.f};
  v8f acc[2][4];
#pragma unroll
  for (int a = 0; a < 2; ++a)
#pragma unroll
    for (int b = 0; b < 4; ++b) acc[a][b] = vzero;

  auto stage = [&](int k0, int buf) {
#pragma unroll
    for (int i = 0; i < 16; ++i) {  // 32x128 tile, coalesced along n
      const int idx = tid + i * 256;
      const int kk  = idx >> 7;
      const int n   = idx & 127;
      Bt[buf][n][kk] = Bm[(size_t)(k0 + kk) * ldb + n0 + n];
    }
  };

  stage(0, 0);
  __syncthreads();

  const int steps = K >> 5;
  for (int t = 0; t < steps; ++t) {
    const int k0  = t << 5;
    const int buf = t & 1;
    if (t + 1 < steps) {
      stage(k0 + 32, buf ^ 1);
      if (t + 2 < steps)  // pull tile-after-next toward the WGP
        __builtin_prefetch(Bm + (size_t)(k0 + 64 + (tid >> 3)) * ldb + n0, 0, 1);
    }

    v16bf a0 = load_a_frag(A, lda, m0, k0, lane);
    v16bf a1 = load_a_frag(A, lda, m0 + 16, k0, lane);
    const int kb = (lane >> 4) << 4;
#pragma unroll
    for (int nt = 0; nt < 4; ++nt) {
      const int nloc = wn * 64 + nt * 16 + (lane & 15);
      u16x16 bv;
#pragma unroll
      for (int i = 0; i < 16; ++i) bv[i] = Bt[buf][nloc][kb + i];
      v16bf bf = __builtin_bit_cast(v16bf, bv);
      acc[0][nt] = wmma_bf16(a0, bf, acc[0][nt]);
      acc[1][nt] = wmma_bf16(a1, bf, acc[1][nt]);
    }
    __syncthreads();  // reads of buf done before it is overwritten at t+2
  }

  const int nco = lane & 15;
  const int mro = (lane >> 4) << 3;
#pragma unroll
  for (int mt = 0; mt < 2; ++mt)
#pragma unroll
    for (int nt = 0; nt < 4; ++nt) {
      const int n = n0 + wn * 64 + nt * 16 + nco;
      const float bb = bias[n];
#pragma unroll
      for (int j = 0; j < 8; ++j) {
        const int m = m0 + mt * 16 + mro + j;
        float v = acc[mt][nt][j] + bb;
        if (RELU) v = fmaxf(v, 0.f);
        if (OUTF) Cf[(size_t)m * ldc + n] = v;
        if (OUTB) Cb[(size_t)m * ldc + n] = f2bf(v);
      }
    }
}

// ------------------------ qE[b,h,q,r] = q . Ek[r] ----------------------------
// One 16x64 tile per wave: M=16 queries, N=64 rel ids, K=64 head dims.

__global__ __launch_bounds__(256) void qe_kernel(const u16* __restrict__ qB,
                                                 const u16* __restrict__ EkB,
                                                 float* __restrict__ qE) {
  const int lane = threadIdx.x & 31;
  const int wave = threadIdx.x >> 5;
  const int z    = blockIdx.x * 8 + wave;  // 0 .. B*H*(S/16)-1
  const int b    = z >> 10;
  const int h    = (z >> 6) & 15;
  const int q0   = (z & 63) << 4;

  const u16* A = qB + (size_t)b * S_ * D_ + h * DH_;
  v16bf a0 = load_a_frag(A, D_, q0, 0, lane);
  v16bf a1 = load_a_frag(A, D_, q0, 32, lane);

  const int mro = (lane >> 4) << 3;
  const v8f vzero = {0.f, 0.f, 0.f, 0.f, 0.f, 0.f, 0.f, 0.f};
#pragma unroll
  for (int nt = 0; nt < 4; ++nt) {
    v16bf b0 = load_bT_frag(EkB, DH_, nt * 16, 0, lane);   // B = Ek^T
    v16bf b1 = load_bT_frag(EkB, DH_, nt * 16, 32, lane);
    v8f acc = vzero;
    acc = wmma_bf16(a0, b0, acc);
    acc = wmma_bf16(a1, b1, acc);
#pragma unroll
    for (int j = 0; j < 8; ++j) {
      const int q = q0 + mro + j;
      const int r = nt * 16 + (lane & 15);
      qE[(((size_t)b * H_ + h) * S_ + q) * R_ + r] = acc[j];
    }
  }
}

// ------------------------------ fused attention ------------------------------
// One block (4 waves) per (b, h, 16-query tile). Scores for the whole key
// axis live in LDS (16x1024 f32 = 64KB) + bf16 probs (32KB). Dynamic LDS is
// used so the 96KB allocation is legal for gfx1250's 320KB/WGP budget.

__global__ __launch_bounds__(128) void attn_kernel(
    const u16* __restrict__ qB, const u16* __restrict__ kB,
    const u16* __restrict__ vT, const float* __restrict__ qE,
    const int* __restrict__ rel, const float* __restrict__ Eb,
    u16* __restrict__ ctxB) {
  extern __shared__ char smem[];
  float* sc = (float*)smem;                 // [16][S_] scores
  u16*   pb = (u16*)(smem + 16 * S_ * 4);   // [16][S_] bf16 probs

  const int lane = threadIdx.x & 31;
  const int wave = threadIdx.x >> 5;
  const int z    = blockIdx.x;
  const int b    = z >> 10;
  const int h    = (z >> 6) & 15;
  const int q0   = (z & 63) << 4;

  const u16* qbase  = qB + (size_t)b * S_ * D_ + h * DH_;
  const u16* kbase  = kB + (size_t)b * S_ * D_ + h * DH_;
  const u16* vTbase = vT + ((size_t)b * H_ + h) * DH_ * S_;  // [dh][s]
  const float* qEb  = qE + ((size_t)b * H_ + h) * S_ * R_;

  v16bf aq0 = load_a_frag(qbase, D_, q0, 0, lane);
  v16bf aq1 = load_a_frag(qbase, D_, q0, 32, lane);

  const int nco = lane & 15;
  const int mro = (lane >> 4) << 3;
  const v8f vzero = {0.f, 0.f, 0.f, 0.f, 0.f, 0.f, 0.f, 0.f};

  // ---- scores: (q.k^T + q.Ek[rid] + Eb[rid,h]) / sqrt(64) ----
  for (int it = 0; it < S_ / 64; ++it) {
    const int nb = it * 64 + wave * 16;
    v16bf bk0 = load_bT_frag(kbase, D_, nb, 0, lane);   // B = K^T, contiguous
    v16bf bk1 = load_bT_frag(kbase, D_, nb, 32, lane);
    v8f s = vzero;
    s = wmma_bf16(aq0, bk0, s);
    s = wmma_bf16(aq1, bk1, s);
#pragma unroll
    for (int j = 0; j < 8; ++j) {
      const int m     = mro + j;
      const int key   = nb + nco;
      const int query = q0 + m;
      const int rid   = rel[(size_t)query * S_ + key];
      sc[m * S_ + key] =
          (s[j] + qEb[(size_t)query * R_ + rid] + Eb[rid * H_ + h]) * 0.125f;
    }
  }
  __syncthreads();

  // ---- softmax: 4 rows per wave, wave-shuffle reductions ----
#pragma unroll
  for (int r = 0; r < 4; ++r) {
    const int row = wave * 4 + r;
    float* p = sc + row * S_;
    float mx = -3.4e38f;
    for (int i = lane; i < S_; i += 32) mx = fmaxf(mx, p[i]);
    mx = wave_max(mx);
    float sum = 0.f;
    for (int i = lane; i < S_; i += 32) {
      const float e = __expf(p[i] - mx);
      p[i] = e;
      sum += e;
    }
    sum = wave_sum(sum);
    const float inv = 1.f / sum;
    u16* pr = pb + row * S_;
    for (int i = lane; i < S_; i += 32) pr[i] = f2bf(p[i] * inv);
  }
  __syncthreads();

  // ---- ctx = P @ V : each wave owns one 16-wide head-dim tile ----
  // V^T rows are contiguous in the key axis -> b128 fragment loads.
  const int n0 = wave * 16;
  v8f acc = vzero;
  for (int k0 = 0; k0 < S_; k0 += 32) {
    v16bf ap = load_a_frag(pb, S_, 0, k0, lane);
    v16bf bv = load_bT_frag(vTbase, S_, n0, k0, lane);
    acc = wmma_bf16(ap, bv, acc);
  }
#pragma unroll
  for (int j = 0; j < 8; ++j) {
    const int m  = mro + j;
    const int dh = n0 + nco;
    ctxB[((size_t)b * S_ + q0 + m) * D_ + h * DH_ + dh] = f2bf(acc[j]);
  }
}

// ------------------------- residual add + LayerNorm --------------------------

__global__ __launch_bounds__(256) void add_ln_kernel(
    const float* __restrict__ x, const float* __restrict__ y,
    const float* __restrict__ g, const float* __restrict__ beta,
    float* __restrict__ outF, u16* __restrict__ outB) {
  __shared__ float red[8];
  const int row  = blockIdx.x;
  const int t    = threadIdx.x;
  const int lane = t & 31;
  const int wave = t >> 5;
  const float* xr = x + (size_t)row * D_;
  const float* yr = y + (size_t)row * D_;

  float v[4];
  float s = 0.f;
#pragma unroll
  for (int i = 0; i < 4; ++i) {
    const int c = t + i * 256;
    v[i] = xr[c] + yr[c];
    s += v[i];
  }
  s = wave_sum(s);
  if (lane == 0) red[wave] = s;
  __syncthreads();
  float mean = 0.f;
#pragma unroll
  for (int i = 0; i < 8; ++i) mean += red[i];
  mean *= (1.f / D_);

  float vs = 0.f;
#pragma unroll
  for (int i = 0; i < 4; ++i) {
    const float d = v[i] - mean;
    vs += d * d;
  }
  __syncthreads();
  vs = wave_sum(vs);
  if (lane == 0) red[wave] = vs;
  __syncthreads();
  float var = 0.f;
#pragma unroll
  for (int i = 0; i < 8; ++i) var += red[i];
  var *= (1.f / D_);
  const float inv = rsqrtf(var + 1e-6f);

#pragma unroll
  for (int i = 0; i < 4; ++i) {
    const int c = t + i * 256;
    const float o = (v[i] - mean) * inv * g[c] + beta[c];
    if (outF) outF[(size_t)row * D_ + c] = o;
    if (outB) outB[(size_t)row * D_ + c] = f2bf(o);
  }
}

// ------------------------------- host driver ---------------------------------

extern "C" void kernel_launch(void* const* d_in, const int* in_sizes, int n_in,
                              void* d_out, int out_size, void* d_ws,
                              size_t ws_size, hipStream_t stream) {
  const float* x   = (const float*)d_in[0];
  const int*   rel = (const int*)d_in[1];
  const float* Wq  = (const float*)d_in[2];
  const float* bq  = (const float*)d_in[3];
  const float* Wk  = (const float*)d_in[4];
  const float* bk  = (const float*)d_in[5];
  const float* Wv  = (const float*)d_in[6];
  const float* bv  = (const float*)d_in[7];
  const float* Wo  = (const float*)d_in[8];
  const float* bo  = (const float*)d_in[9];
  const float* Ek  = (const float*)d_in[10];
  const float* Eb  = (const float*)d_in[11];
  const float* g1  = (const float*)d_in[12];
  const float* b1  = (const float*)d_in[13];
  const float* g2  = (const float*)d_in[14];
  const float* b2  = (const float*)d_in[15];
  const float* W1  = (const float*)d_in[16];
  const float* bf1 = (const float*)d_in[17];
  const float* W2  = (const float*)d_in[18];
  const float* bf2 = (const float*)d_in[19];
  float* out = (float*)d_out;

  char* wsp = (char*)d_ws;
  size_t off = 0;
  auto take = [&](size_t bytes) -> char* {
    char* p = wsp + off;
    off += (bytes + 255) & ~(size_t)255;
    return p;
  };

  u16* xB    = (u16*)take((size_t)TOK_ * D_ * 2);
  u16* WqB   = (u16*)take((size_t)D_ * D_ * 2);
  u16* WkB   = (u16*)take((size_t)D_ * D_ * 2);
  u16* WvB   = (u16*)take((size_t)D_ * D_ * 2);
  u16* WoB   = (u16*)take((size_t)D_ * D_ * 2);
  u16* W1B   = (u16*)take((size_t)D_ * FFN_ * 2);
  u16* W2B   = (u16*)take((size_t)FFN_ * D_ * 2);
  u16* EkB   = (u16*)take((size_t)R_ * DH_ * 2);
  u16* qBuf  = (u16*)take((size_t)TOK_ * D_ * 2);
  u16* kBuf  = (u16*)take((size_t)TOK_ * D_ * 2);
  u16* vBuf  = (u16*)take((size_t)TOK_ * D_ * 2);
  u16* vTB   = (u16*)take((size_t)TOK_ * D_ * 2);
  float* qE  = (float*)take((size_t)B_ * H_ * S_ * R_ * 4);
  u16* ctxB  = (u16*)take((size_t)TOK_ * D_ * 2);
  float* aout  = (float*)take((size_t)TOK_ * D_ * 4);
  float* ffin  = (float*)take((size_t)TOK_ * D_ * 4);
  u16* ffinB   = (u16*)take((size_t)TOK_ * D_ * 2);
  u16* h1B     = (u16*)take((size_t)TOK_ * FFN_ * 2);
  float* ffout = (float*)take((size_t)TOK_ * D_ * 4);

  auto cvt = [&](const float* s, u16* d, int n) {
    cvt_kernel<<<512, 256, 0, stream>>>(s, d, n);
  };
  cvt(x, xB, TOK_ * D_);
  cvt(Wq, WqB, D_ * D_);
  cvt(Wk, WkB, D_ * D_);
  cvt(Wv, WvB, D_ * D_);
  cvt(Wo, WoB, D_ * D_);
  cvt(W1, W1B, D_ * FFN_);
  cvt(W2, W2B, FFN_ * D_);
  cvt(Ek, EkB, R_ * DH_);

  // Q,K,V projections -> bf16 activations (bias, no relu, bf16 out)
  {
    dim3 grid(D_ / 128, TOK_ / 128);
    gemm_bf16_kernel<false, false, true><<<grid, 256, 0, stream>>>(
        xB, D_, WqB, D_, bq, nullptr, qBuf, D_, D_);
    gemm_bf16_kernel<false, false, true><<<grid, 256, 0, stream>>>(
        xB, D_, WkB, D_, bk, nullptr, kBuf, D_, D_);
    gemm_bf16_kernel<false, false, true><<<grid, 256, 0, stream>>>(
        xB, D_, WvB, D_, bv, nullptr, vBuf, D_, D_);
  }

  // V transpose for contiguous context-GEMM fragments
  vtrans_kernel<<<4096, 256, 0, stream>>>(vBuf, vTB);

  // qE = q . Ek^T   (per b,h; 16-query tile per wave)
  qe_kernel<<<(B_ * H_ * (S_ / 16)) / 8, 256, 0, stream>>>(qBuf, EkB, qE);

  // fused relative attention
  attn_kernel<<<B_ * H_ * (S_ / 16), 128, 16 * S_ * 4 + 16 * S_ * 2, stream>>>(
      qBuf, kBuf, vTB, qE, rel, Eb, ctxB);

  // output projection (f32 out)
  {
    dim3 grid(D_ / 128, TOK_ / 128);
    gemm_bf16_kernel<false, true, false><<<grid, 256, 0, stream>>>(
        ctxB, D_, WoB, D_, bo, aout, nullptr, D_, D_);
  }

  // ff_in = LN(x + a_out)
  add_ln_kernel<<<TOK_, 256, 0, stream>>>(x, aout, g1, b1, ffin, ffinB);

  // FFN
  {
    dim3 g1d(FFN_ / 128, TOK_ / 128);
    gemm_bf16_kernel<true, false, true><<<g1d, 256, 0, stream>>>(
        ffinB, D_, W1B, FFN_, bf1, nullptr, h1B, FFN_, D_);
    dim3 g2d(D_ / 128, TOK_ / 128);
    gemm_bf16_kernel<false, true, false><<<g2d, 256, 0, stream>>>(
        h1B, FFN_, W2B, D_, bf2, ffout, nullptr, D_, FFN_);
  }

  // out = LN(ff_in + ff)
  add_ln_kernel<<<TOK_, 256, 0, stream>>>(ffin, ffout, g2, b2, out, nullptr);

  (void)in_sizes; (void)n_in; (void)out_size; (void)ws_size;
}
